// JitPaiNN_5076651344266
// MI455X (gfx1250) — matrix operand order repaired and checked by reading, not compiled, not run on hardware
//
#include <hip/hip_runtime.h>
#include <math.h>

#define NA 8192
#define NE 131072
#define FD 128
#define F3 384
#define NBASIS 20
#define NBPAD 32
#define HD 64
#define RC 5.0f
#define PI_F 3.14159265358979323846f

typedef __attribute__((ext_vector_type(16))) __bf16 v16bf;
typedef __attribute__((ext_vector_type(8)))  __bf16 v8bf;
typedef __attribute__((ext_vector_type(8)))  float  v8f;

#define GF_ACCUM 1

static __device__ __forceinline__ void put4(v16bf& v, int i, float4 f) {
  v[i + 0] = (__bf16)f.x; v[i + 1] = (__bf16)f.y;
  v[i + 2] = (__bf16)f.z; v[i + 3] = (__bf16)f.w;
}
static __device__ __forceinline__ float siluf(float x) {
  float s = 1.f / (1.f + __expf(-x));
  return x * s;
}
static __device__ __forceinline__ float dsiluf(float x) {
  float s = 1.f / (1.f + __expf(-x));
  return s * (1.f + x * (1.f - s));
}

// ---------------- bf16-WMMA GEMM: C[M,Nc] = A[M,K](f32) * Bt^T(bf16) (+bias) --
// Bt stored transposed in bf16: Bt[n*ldb + k] (K contiguous). K % 32 == 0.
// One wave32 per block computing a 16x64 C strip (4 WMMA col tiles, A reused).
__global__ void __launch_bounds__(32)
k_gemm(const float* __restrict__ A, int lda,
       const __bf16* __restrict__ Bt, int ldb,
       float* __restrict__ C, int ldc,
       int M, int Nc, int K,
       const float* __restrict__ bias, int flags) {
  const int lane = threadIdx.x & 31;
  const int half = lane >> 4;
  const int ln   = lane & 15;
  const int r0   = blockIdx.y << 4;
  const int c0   = blockIdx.x << 6;
  const float* arow = A + (size_t)(r0 + ln) * (size_t)lda;

  v8f acc[4];
  const __bf16* brow[4];
#pragma unroll
  for (int s = 0; s < 4; ++s) {
    acc[s] = (v8f){0.f,0.f,0.f,0.f,0.f,0.f,0.f,0.f};
    int c = c0 + (s << 4) + ln;
    int ce = (c < Nc) ? c : (Nc - 1);          // branchless clamp: loads stay valid
    brow[s] = Bt + (size_t)ce * (size_t)ldb;
  }

  for (int k0 = 0; k0 < K; k0 += 32) {
    __builtin_prefetch(arow + k0 + 64, 0, 1);
    // A fragment: lanes0-15 K={0..7,16..23}, lanes16-31 K={8..15,24..31}
    const float4* ap0 = (const float4*)(arow + k0 + 8 * half);
    const float4* ap1 = (const float4*)(arow + k0 + 16 + 8 * half);
    float4 aA = ap0[0], aB = ap0[1], aC = ap1[0], aD = ap1[1];
    v16bf af;
    put4(af, 0, aA); put4(af, 4, aB); put4(af, 8, aC); put4(af, 12, aD);
#pragma unroll
    for (int s = 0; s < 4; ++s) {
      const v8bf* bp = (const v8bf*)(brow[s] + k0 + 8 * half);
      const v8bf* bq = (const v8bf*)(brow[s] + k0 + 16 + 8 * half);
      v8bf blo = bp[0], bhi = bq[0];
      v16bf bfv = __builtin_shufflevector(blo, bhi,
                    0,1,2,3,4,5,6,7,8,9,10,11,12,13,14,15);
      acc[s] = __builtin_amdgcn_wmma_f32_16x16x32_bf16(
          false, af, false, bfv, (short)0, acc[s], false, false);
    }
  }
#pragma unroll
  for (int s = 0; s < 4; ++s) {
    int col = c0 + (s << 4) + ln;
    if (col >= Nc) continue;
    float bb = bias ? bias[col] : 0.f;
#pragma unroll
    for (int j = 0; j < 8; ++j) {
      size_t idx = (size_t)(r0 + j + 8 * half) * (size_t)ldc + col;
      float v = acc[s][j] + bb;
      if (flags & GF_ACCUM) v += C[idx];
      C[idx] = v;
    }
  }
}

// ---------------- weight prep (fp32 -> bf16) ----------------------------------
// src[R,C] -> dst[C,Rp] transposed: dst[c*Rp+r] = (r<R) ? src[r*C+c] : 0
__global__ void k_transpose_bf16(__bf16* __restrict__ dst, const float* __restrict__ src,
                                 int R, int C, int Rp) {
  long t = (long)blockIdx.x * blockDim.x + threadIdx.x;
  if (t >= (long)C * Rp) return;
  int c = (int)(t / Rp), r = (int)(t % Rp);
  dst[(size_t)c * Rp + r] = (__bf16)((r < R) ? src[(size_t)r * C + c] : 0.f);
}
// src[R,C] -> dst[Rp,C] bf16, zero-padding rows R..Rp-1
__global__ void k_pad_rows_bf16(__bf16* __restrict__ dst, const float* __restrict__ src,
                                int R, int C, int Rp) {
  long t = (long)blockIdx.x * blockDim.x + threadIdx.x;
  if (t >= (long)Rp * C) return;
  int r = (int)(t / C);
  dst[t] = (__bf16)((r < R) ? src[t] : 0.f);
}
// identity-layout cast
__global__ void k_cast_bf16(__bf16* __restrict__ dst, const float* __restrict__ src, long n) {
  long t = (long)blockIdx.x * blockDim.x + threadIdx.x;
  if (t < n) dst[t] = (__bf16)src[t];
}

// ---------------- elementwise helpers ----------------------------------------
__global__ void k_zero(float* __restrict__ p, long n) {
  long t = (long)blockIdx.x * blockDim.x + threadIdx.x;
  if (t < n) p[t] = 0.f;
}
__global__ void k_copy(float* __restrict__ d, const float* __restrict__ s, long n) {
  long t = (long)blockIdx.x * blockDim.x + threadIdx.x;
  if (t < n) d[t] = s[t];
}
__global__ void k_silu(const float* __restrict__ t, float* __restrict__ s, long n) {
  long i = (long)blockIdx.x * blockDim.x + threadIdx.x;
  if (i < n) s[i] = siluf(t[i]);
}
__global__ void k_silu_bwd(const float* __restrict__ gs, const float* __restrict__ t,
                           float* __restrict__ gt, long n) {
  long i = (long)blockIdx.x * blockDim.x + threadIdx.x;
  if (i < n) gt[i] = gs[i] * dsiluf(t[i]);
}
__global__ void k_embed(const int* __restrict__ at_no, const float* __restrict__ emb,
                        float* __restrict__ x0) {
  long t = (long)blockIdx.x * blockDim.x + threadIdx.x;
  if (t >= (long)NA * FD) return;
  int n = (int)(t >> 7), f = (int)(t & 127);
  x0[t] = emb[(size_t)at_no[n] * FD + f];
}

// ---------------- edge geometry ----------------------------------------------
__global__ void k_edge_geom(const float* __restrict__ coord, const int* __restrict__ eidx,
                            float* __restrict__ dist, float* __restrict__ unitv,
                            float* __restrict__ rbf, float* __restrict__ fcut) {
  int e = blockIdx.x * blockDim.x + threadIdx.x;
  if (e >= NE) return;
  int s = eidx[e], t = eidx[NE + e];
  float rx = coord[t*3+0] - coord[s*3+0];
  float ry = coord[t*3+1] - coord[s*3+1];
  float rz = coord[t*3+2] - coord[s*3+2];
  float d = sqrtf(rx*rx + ry*ry + rz*rz + 1e-12f);
  float inv = 1.f / d;
  unitv[e*3+0] = rx*inv; unitv[e*3+1] = ry*inv; unitv[e*3+2] = rz*inv;
  float coef = sqrtf(2.f / RC);
  float w = PI_F * d / RC;
#pragma unroll
  for (int k = 0; k < NBASIS; ++k)
    rbf[(size_t)e*NBPAD + k] = coef * __sinf((float)(k+1) * w) * inv;
#pragma unroll
  for (int k = NBASIS; k < NBPAD; ++k)
    rbf[(size_t)e*NBPAD + k] = 0.f;
  fcut[e] = (d < RC) ? 0.5f * (__cosf(w) + 1.f) : 0.f;
  dist[e] = d;
}

// ---------------- message forward: m = phi[src]*filt, scatter to dst ---------
__global__ void k_edge_msg_fwd(const float* __restrict__ phi, const float* __restrict__ fr,
                               const float* __restrict__ fcut, const float* __restrict__ unitv,
                               const float* __restrict__ v_in, const int* __restrict__ eidx,
                               float* __restrict__ x_acc, float* __restrict__ v_acc, int vzero) {
  long tid = (long)blockIdx.x * blockDim.x + threadIdx.x;
  int e = (int)(tid >> 7), f = (int)(tid & 127);
  if (e >= NE) return;
  int s = eidx[e], t = eidx[NE + e];
  float fc = fcut[e];
  const float* pr = phi + (size_t)s * F3;
  const float* fe = fr + (size_t)e * F3;
  float ds  = pr[f]       * fe[f]       * fc;
  float dv1 = pr[128 + f] * fe[128 + f] * fc;
  float dv2 = pr[256 + f] * fe[256 + f] * fc;
  atomicAdd(&x_acc[(size_t)t * FD + f], ds);
#pragma unroll
  for (int i = 0; i < 3; ++i) {
    float u  = unitv[e*3 + i];
    float vi = vzero ? 0.f : v_in[((size_t)s*3 + i) * FD + f];
    atomicAdd(&v_acc[((size_t)t*3 + i) * FD + f], dv1 * vi + dv2 * u);
  }
}

// ---------------- update block forward helpers --------------------------------
__global__ void k_vnorm_cat(const float* __restrict__ vv, const float* __restrict__ xacc,
                            float* __restrict__ vnorm, float* __restrict__ cat) {
  long t = (long)blockIdx.x * blockDim.x + threadIdx.x;
  if (t >= (long)NA * FD) return;
  int n = (int)(t >> 7), f = (int)(t & 127);
  float a0 = vv[((size_t)n*3 + 0)*FD + f];
  float a1 = vv[((size_t)n*3 + 1)*FD + f];
  float a2 = vv[((size_t)n*3 + 2)*FD + f];
  float vn = sqrtf(a0*a0 + a1*a1 + a2*a2 + 1e-12f);
  vnorm[t] = vn;
  cat[(size_t)n*256 + f]       = xacc[t];
  cat[(size_t)n*256 + 128 + f] = vn;
}
__global__ void k_update_apply(const float* __restrict__ uv, const float* __restrict__ vv,
                               const float* __restrict__ a,
                               float* __restrict__ xacc, float* __restrict__ vacc) {
  long t = (long)blockIdx.x * blockDim.x + threadIdx.x;
  if (t >= (long)NA * FD) return;
  int n = (int)(t >> 7), f = (int)(t & 127);
  float u0 = uv[((size_t)n*3+0)*FD+f], u1 = uv[((size_t)n*3+1)*FD+f], u2 = uv[((size_t)n*3+2)*FD+f];
  float w0 = vv[((size_t)n*3+0)*FD+f], w1 = vv[((size_t)n*3+1)*FD+f], w2 = vv[((size_t)n*3+2)*FD+f];
  float sc  = u0*w0 + u1*w1 + u2*w2;
  float avv = a[(size_t)n*F3 + f];
  float asv = a[(size_t)n*F3 + 128 + f];
  float ass = a[(size_t)n*F3 + 256 + f];
  xacc[t] += asv * sc + ass;
  vacc[((size_t)n*3+0)*FD+f] += avv * u0;
  vacc[((size_t)n*3+1)*FD+f] += avv * u1;
  vacc[((size_t)n*3+2)*FD+f] += avv * u2;
}

// ---------------- output head: energy reduction + g_t3 ------------------------
__global__ void k_energy(const float* __restrict__ t3, const float* __restrict__ w2,
                         const float* __restrict__ b2, const int* __restrict__ at_no,
                         const float* __restrict__ atom_sp, float* __restrict__ out) {
  int n = blockIdx.x * blockDim.x + threadIdx.x;
  float acc = 0.f;
  if (n < NA) {
#pragma unroll 8
    for (int h = 0; h < HD; ++h) acc += siluf(t3[(size_t)n*HD + h]) * w2[h];
    acc += b2[0] + atom_sp[at_no[n]];
  }
#pragma unroll
  for (int off = 16; off; off >>= 1) acc += __shfl_down(acc, off);
  if ((threadIdx.x & 31) == 0) atomicAdd(&out[0], acc);
}
__global__ void k_gt3(const float* __restrict__ t3, const float* __restrict__ w2,
                      float* __restrict__ gt3) {
  long t = (long)blockIdx.x * blockDim.x + threadIdx.x;
  if (t >= (long)NA * HD) return;
  int h = (int)(t % HD);
  gt3[t] = w2[h] * dsiluf(t3[t]);
}

// ---------------- update block backward --------------------------------------
__global__ void k_upd_bwd1(const float* __restrict__ gx, const float* __restrict__ gv,
                           const float* __restrict__ uv, const float* __restrict__ vv,
                           const float* __restrict__ a,
                           float* __restrict__ ga, float* __restrict__ guv,
                           float* __restrict__ gvv) {
  long t = (long)blockIdx.x * blockDim.x + threadIdx.x;
  if (t >= (long)NA * FD) return;
  int n = (int)(t >> 7), f = (int)(t & 127);
  float u0 = uv[((size_t)n*3+0)*FD+f], u1 = uv[((size_t)n*3+1)*FD+f], u2 = uv[((size_t)n*3+2)*FD+f];
  float w0 = vv[((size_t)n*3+0)*FD+f], w1 = vv[((size_t)n*3+1)*FD+f], w2 = vv[((size_t)n*3+2)*FD+f];
  float g0 = gv[((size_t)n*3+0)*FD+f], g1 = gv[((size_t)n*3+1)*FD+f], g2 = gv[((size_t)n*3+2)*FD+f];
  float sc  = u0*w0 + u1*w1 + u2*w2;
  float g   = gx[t];
  float avv = a[(size_t)n*F3 + f];
  float asv = a[(size_t)n*F3 + 128 + f];
  float gsc = g * asv;
  ga[(size_t)n*F3 + f]       = g0*u0 + g1*u1 + g2*u2;   // g_a_vv
  ga[(size_t)n*F3 + 128 + f] = g * sc;                  // g_a_sv
  ga[(size_t)n*F3 + 256 + f] = g;                       // g_a_ss
  guv[((size_t)n*3+0)*FD+f] = g0*avv + gsc*w0;
  guv[((size_t)n*3+1)*FD+f] = g1*avv + gsc*w1;
  guv[((size_t)n*3+2)*FD+f] = g2*avv + gsc*w2;
  gvv[((size_t)n*3+0)*FD+f] = gsc*u0;
  gvv[((size_t)n*3+1)*FD+f] = gsc*u1;
  gvv[((size_t)n*3+2)*FD+f] = gsc*u2;
}
__global__ void k_upd_bwd2(const float* __restrict__ gcat, const float* __restrict__ vnorm,
                           const float* __restrict__ vv,
                           float* __restrict__ gx, float* __restrict__ gvv) {
  long t = (long)blockIdx.x * blockDim.x + threadIdx.x;
  if (t >= (long)NA * FD) return;
  int n = (int)(t >> 7), f = (int)(t & 127);
  gx[t] += gcat[(size_t)n*256 + f];
  float gvn = gcat[(size_t)n*256 + 128 + f] / vnorm[t];
  gvv[((size_t)n*3+0)*FD+f] += gvn * vv[((size_t)n*3+0)*FD+f];
  gvv[((size_t)n*3+1)*FD+f] += gvn * vv[((size_t)n*3+1)*FD+f];
  gvv[((size_t)n*3+2)*FD+f] += gvn * vv[((size_t)n*3+2)*FD+f];
}

// ---------------- message block backward (edge pass) --------------------------
__global__ void k_edge_msg_bwd(const float* __restrict__ phi, float* __restrict__ fr,
                               const float* __restrict__ fcut, const float* __restrict__ unitv,
                               const float* __restrict__ v_in, const int* __restrict__ eidx,
                               const float* __restrict__ gx, const float* __restrict__ gvp,
                               float* __restrict__ gv, float* __restrict__ gphi,
                               float* __restrict__ gunit, float* __restrict__ gfcut, int vzero) {
  long tid = (long)blockIdx.x * blockDim.x + threadIdx.x;
  int e = (int)(tid >> 7), f = (int)(tid & 127);
  if (e >= NE) return;
  int s = eidx[e], t = eidx[NE + e];
  float fc = fcut[e];
  const float* pr = phi + (size_t)s * F3;
  float p0 = pr[f], p1 = pr[128 + f], p2 = pr[256 + f];
  size_t fbase = (size_t)e * F3;
  float r0 = fr[fbase + f], r1 = fr[fbase + 128 + f], r2 = fr[fbase + 256 + f];
  float gds = gx[(size_t)t * FD + f];
  float gd0 = gvp[((size_t)t*3+0)*FD+f];
  float gd1 = gvp[((size_t)t*3+1)*FD+f];
  float gd2 = gvp[((size_t)t*3+2)*FD+f];
  float u0 = unitv[e*3+0], u1 = unitv[e*3+1], u2 = unitv[e*3+2];
  float vi0 = vzero ? 0.f : v_in[((size_t)s*3+0)*FD+f];
  float vi1 = vzero ? 0.f : v_in[((size_t)s*3+1)*FD+f];
  float vi2 = vzero ? 0.f : v_in[((size_t)s*3+2)*FD+f];
  float gdv1 = gd0*vi0 + gd1*vi1 + gd2*vi2;
  float gdv2 = gd0*u0 + gd1*u1 + gd2*u2;
  atomicAdd(&gphi[(size_t)s*F3 + f],       gds  * r0 * fc);
  atomicAdd(&gphi[(size_t)s*F3 + 128 + f], gdv1 * r1 * fc);
  atomicAdd(&gphi[(size_t)s*F3 + 256 + f], gdv2 * r2 * fc);
  if (!vzero) {
    float dv1 = p1 * r1 * fc;
    atomicAdd(&gv[((size_t)s*3+0)*FD+f], gd0 * dv1);
    atomicAdd(&gv[((size_t)s*3+1)*FD+f], gd1 * dv1);
    atomicAdd(&gv[((size_t)s*3+2)*FD+f], gd2 * dv1);
  }
  float dv2 = p2 * r2 * fc;
  float gu0 = gd0 * dv2, gu1 = gd1 * dv2, gu2 = gd2 * dv2;
  float gfc = gds * p0 * r0 + gdv1 * p1 * r1 + gdv2 * p2 * r2;
  fr[fbase + f]       = gds  * p0 * fc;          // overwrite with g_filt_raw
  fr[fbase + 128 + f] = gdv1 * p1 * fc;
  fr[fbase + 256 + f] = gdv2 * p2 * fc;
#pragma unroll
  for (int off = 16; off; off >>= 1) {
    gu0 += __shfl_down(gu0, off);
    gu1 += __shfl_down(gu1, off);
    gu2 += __shfl_down(gu2, off);
    gfc += __shfl_down(gfc, off);
  }
  if ((threadIdx.x & 31) == 0) {
    atomicAdd(&gunit[e*3+0], gu0);
    atomicAdd(&gunit[e*3+1], gu1);
    atomicAdd(&gunit[e*3+2], gu2);
    atomicAdd(&gfcut[e], gfc);
  }
}

// ---------------- geometry backward -------------------------------------------
__global__ void k_geom_bwd(const float* __restrict__ dist, const float* __restrict__ unitv,
                           const int* __restrict__ eidx,
                           const float* __restrict__ grbf, const float* __restrict__ gfcut,
                           const float* __restrict__ gunit, float* __restrict__ gcoord) {
  int e = blockIdx.x * blockDim.x + threadIdx.x;
  if (e >= NE) return;
  float d = dist[e];
  float invd = 1.f / d;
  float gd = (d < RC) ? gfcut[e] * (-0.5f * PI_F / RC * __sinf(PI_F * d / RC)) : 0.f;
  float coef = sqrtf(2.f / RC);
#pragma unroll
  for (int k = 0; k < NBASIS; ++k) {
    float nk = (float)(k + 1);
    float arg = nk * PI_F * d / RC;
    float sn = __sinf(arg), cs = __cosf(arg);
    float drbf = coef * (nk * PI_F / RC * cs * invd - sn * invd * invd);
    gd += grbf[(size_t)e*NBPAD + k] * drbf;
  }
  float u0 = unitv[e*3+0], u1 = unitv[e*3+1], u2 = unitv[e*3+2];
  float g0 = gunit[e*3+0], g1 = gunit[e*3+1], g2 = gunit[e*3+2];
  float dot = g0*u0 + g1*u1 + g2*u2;
  float gr0 = g0*invd - dot*u0*invd + gd*u0;
  float gr1 = g1*invd - dot*u1*invd + gd*u1;
  float gr2 = g2*invd - dot*u2*invd + gd*u2;
  int s = eidx[e], t = eidx[NE + e];
  atomicAdd(&gcoord[t*3+0],  gr0); atomicAdd(&gcoord[t*3+1],  gr1); atomicAdd(&gcoord[t*3+2],  gr2);
  atomicAdd(&gcoord[s*3+0], -gr0); atomicAdd(&gcoord[s*3+1], -gr1); atomicAdd(&gcoord[s*3+2], -gr2);
}

// =============================================================================
extern "C" void kernel_launch(void* const* d_in, const int* in_sizes, int n_in,
                              void* d_out, int out_size, void* d_ws, size_t ws_size,
                              hipStream_t stream) {
  (void)in_sizes; (void)n_in; (void)out_size; (void)ws_size;
  const int*   at_no   = (const int*)d_in[0];
  const float* coord   = (const float*)d_in[1];
  const int*   eidx    = (const int*)d_in[2];
  const float* emb     = (const float*)d_in[5];
  const float* atom_sp = (const float*)d_in[6];
  const float* msg_w1  = (const float*)d_in[7];
  const float* msg_b1  = (const float*)d_in[8];
  const float* msg_w2  = (const float*)d_in[9];
  const float* msg_b2  = (const float*)d_in[10];
  const float* msg_wr  = (const float*)d_in[11];
  const float* msg_br  = (const float*)d_in[12];
  const float* upd_u   = (const float*)d_in[13];
  const float* upd_v   = (const float*)d_in[14];
  const float* upd_w1  = (const float*)d_in[15];
  const float* upd_b1  = (const float*)d_in[16];
  const float* upd_w2  = (const float*)d_in[17];
  const float* upd_b2  = (const float*)d_in[18];
  const float* out_w1  = (const float*)d_in[19];
  const float* out_b1  = (const float*)d_in[20];
  const float* out_w2  = (const float*)d_in[21];
  const float* out_b2  = (const float*)d_in[22];
  float* out = (float*)d_out;

  const size_t NF = (size_t)NA * FD;
  float* ws = (float*)d_ws;
  size_t off = 0;
  auto alloc  = [&](size_t n) { float* p = ws + off; off += n; return p; };
  auto allocb = [&](size_t n) { __bf16* p = (__bf16*)(ws + off); off += (n + 1) / 2; return p; };

  float* dbuf  = alloc(NE);
  float* fcutb = alloc(NE);
  float* unitb = alloc(3 * (size_t)NE);
  float* rbfb  = alloc((size_t)NBPAD * NE);
  float* x0    = alloc(NF);
  float* xacc[2] = {alloc(NF), alloc(NF)};
  float* vacc[2] = {alloc(3*NF), alloc(3*NF)};
  float* t1b[2]  = {alloc(NF), alloc(NF)};
  float* phib[2] = {alloc(3*NF), alloc(3*NF)};
  float* uvb[2]  = {alloc(3*NF), alloc(3*NF)};
  float* vvb[2]  = {alloc(3*NF), alloc(3*NF)};
  float* t2b[2]  = {alloc(NF), alloc(NF)};
  float* ab[2]   = {alloc(3*NF), alloc(3*NF)};
  float* vnb[2]  = {alloc(NF), alloc(NF)};
  float* catb  = alloc(2*NF);
  float* t3b   = alloc((size_t)NA * HD);
  float* gt3   = alloc((size_t)NA * HD);
  float* TA    = alloc(3*NF);
  float* TB    = alloc(3*NF);
  float* gx    = alloc(NF);
  float* gv    = alloc(3*NF);
  float* gphi  = alloc(3*NF);
  float* guv   = alloc(3*NF);
  float* gvv   = alloc(3*NF);
  float* grbf  = alloc((size_t)NBPAD * NE);
  float* gfcut = alloc(NE);
  float* gunit = alloc(3 * (size_t)NE);
  // bf16 weights: forward transposed Bt, backward identity-layout Bt (= W)
  __bf16* W1t[2]  = {allocb(FD*FD),    allocb(FD*FD)};
  __bf16* W2t[2]  = {allocb(F3*FD),    allocb(F3*FD)};
  __bf16* Wrt[2]  = {allocb(F3*NBPAD), allocb(F3*NBPAD)};
  __bf16* Utt[2]  = {allocb(FD*FD),    allocb(FD*FD)};
  __bf16* Vtt[2]  = {allocb(FD*FD),    allocb(FD*FD)};
  __bf16* Wu1t[2] = {allocb(FD*2*FD),  allocb(FD*2*FD)};
  __bf16* Wu2t[2] = {allocb(F3*FD),    allocb(F3*FD)};
  __bf16* Wrp[2]  = {allocb(NBPAD*F3), allocb(NBPAD*F3)};
  __bf16* Wo1t    = allocb(HD*FD);
  __bf16* W1c[2]  = {allocb(FD*FD),    allocb(FD*FD)};
  __bf16* W2c[2]  = {allocb(FD*F3),    allocb(FD*F3)};
  __bf16* Uc[2]   = {allocb(FD*FD),    allocb(FD*FD)};
  __bf16* Vc[2]   = {allocb(FD*FD),    allocb(FD*FD)};
  __bf16* Wu1c[2] = {allocb(2*FD*FD),  allocb(2*FD*FD)};
  __bf16* Wu2c[2] = {allocb(FD*F3),    allocb(FD*F3)};
  __bf16* Wo1c    = allocb(FD*HD);
  float* filtb = alloc((size_t)NE * F3);

  auto gemm = [&](const float* A, int lda, const __bf16* Bt, int ldb, float* C, int ldc,
                  int M, int Nc, int K, const float* bias, int flags) {
    dim3 g((Nc + 63) / 64, M / 16);
    k_gemm<<<g, 32, 0, stream>>>(A, lda, Bt, ldb, C, ldc, M, Nc, K, bias, flags);
  };
  auto zero = [&](float* p, long n) { k_zero<<<(unsigned)((n + 255) / 256), 256, 0, stream>>>(p, n); };
  auto copy = [&](float* d, const float* s, long n) { k_copy<<<(unsigned)((n + 255) / 256), 256, 0, stream>>>(d, s, n); };
  auto tpose = [&](__bf16* d, const float* s, int R, int C, int Rp) {
    long n = (long)C * Rp;
    k_transpose_bf16<<<(unsigned)((n + 255) / 256), 256, 0, stream>>>(d, s, R, C, Rp);
  };
  auto castb = [&](__bf16* d, const float* s, long n) {
    k_cast_bf16<<<(unsigned)((n + 255) / 256), 256, 0, stream>>>(d, s, n);
  };
  const unsigned EFB = (unsigned)(((long)NE * FD) / 256);
  const unsigned NFB = (unsigned)((NF + 255) / 256);

  // ---------------- weight prep ----------------
  for (int b = 0; b < 2; ++b) {
    tpose(W1t[b],  msg_w1 + (size_t)b*FD*FD,     FD,     FD, FD);
    tpose(W2t[b],  msg_w2 + (size_t)b*FD*F3,     FD,     F3, FD);
    tpose(Wrt[b],  msg_wr + (size_t)b*NBASIS*F3, NBASIS, F3, NBPAD);
    tpose(Utt[b],  upd_u  + (size_t)b*FD*FD,     FD,     FD, FD);
    tpose(Vtt[b],  upd_v  + (size_t)b*FD*FD,     FD,     FD, FD);
    tpose(Wu1t[b], upd_w1 + (size_t)b*2*FD*FD,   2*FD,   FD, 2*FD);
    tpose(Wu2t[b], upd_w2 + (size_t)b*FD*F3,     FD,     F3, FD);
    k_pad_rows_bf16<<<(NBPAD*F3 + 255)/256, 256, 0, stream>>>(
        Wrp[b], msg_wr + (size_t)b*NBASIS*F3, NBASIS, F3, NBPAD);
    castb(W1c[b],  msg_w1 + (size_t)b*FD*FD,   FD*FD);
    castb(W2c[b],  msg_w2 + (size_t)b*FD*F3,   (long)FD*F3);
    castb(Uc[b],   upd_u  + (size_t)b*FD*FD,   FD*FD);
    castb(Vc[b],   upd_v  + (size_t)b*FD*FD,   FD*FD);
    castb(Wu1c[b], upd_w1 + (size_t)b*2*FD*FD, (long)2*FD*FD);
    castb(Wu2c[b], upd_w2 + (size_t)b*FD*F3,   (long)FD*F3);
  }
  tpose(Wo1t, out_w1, FD, HD, FD);
  castb(Wo1c, out_w1, (long)FD*HD);

  // ---------------- forward ----------------
  zero(out, 1 + 3L * NA);
  k_edge_geom<<<(NE + 255) / 256, 256, 0, stream>>>(coord, eidx, dbuf, unitb, rbfb, fcutb);
  k_embed<<<NFB, 256, 0, stream>>>(at_no, emb, x0);

  for (int b = 0; b < 2; ++b) {
    const float* B1  = msg_b1 + (size_t)b * FD;
    const float* B2  = msg_b2 + (size_t)b * F3;
    const float* Br  = msg_br + (size_t)b * F3;
    const float* Bu1 = upd_b1 + (size_t)b * FD;
    const float* Bu2 = upd_b2 + (size_t)b * F3;
    const float* x_in = (b == 0) ? x0 : xacc[0];

    gemm(x_in, FD, W1t[b], FD, t1b[b], FD, NA, FD, FD, B1, 0);
    k_silu<<<NFB, 256, 0, stream>>>(t1b[b], TA, NF);
    gemm(TA, FD, W2t[b], FD, phib[b], F3, NA, F3, FD, B2, 0);
    gemm(rbfb, NBPAD, Wrt[b], NBPAD, filtb, F3, NE, F3, NBPAD, Br, 0);

    copy(xacc[b], x_in, NF);
    if (b == 0) zero(vacc[0], 3 * NF); else copy(vacc[1], vacc[0], 3 * NF);
    k_edge_msg_fwd<<<EFB, 256, 0, stream>>>(phib[b], filtb, fcutb, unitb,
                                            vacc[0], eidx, xacc[b], vacc[b], b == 0 ? 1 : 0);
    gemm(vacc[b], FD, Utt[b], FD, uvb[b], FD, 3 * NA, FD, FD, nullptr, 0);
    gemm(vacc[b], FD, Vtt[b], FD, vvb[b], FD, 3 * NA, FD, FD, nullptr, 0);
    k_vnorm_cat<<<NFB, 256, 0, stream>>>(vvb[b], xacc[b], vnb[b], catb);
    gemm(catb, 2 * FD, Wu1t[b], 2 * FD, t2b[b], FD, NA, FD, 2 * FD, Bu1, 0);
    k_silu<<<NFB, 256, 0, stream>>>(t2b[b], TA, NF);
    gemm(TA, FD, Wu2t[b], FD, ab[b], F3, NA, F3, FD, Bu2, 0);
    k_update_apply<<<NFB, 256, 0, stream>>>(uvb[b], vvb[b], ab[b], xacc[b], vacc[b]);
  }
  gemm(xacc[1], FD, Wo1t, FD, t3b, HD, NA, HD, FD, out_b1, 0);
  k_energy<<<NA / 256, 256, 0, stream>>>(t3b, out_w2, out_b2, at_no, atom_sp, out);

  // ---------------- backward ----------------
  k_gt3<<<(unsigned)(((size_t)NA * HD + 255) / 256), 256, 0, stream>>>(t3b, out_w2, gt3);
  gemm(gt3, HD, Wo1c, HD, gx, FD, NA, FD, HD, nullptr, 0);   // Wo1c[F,H] = Bt
  zero(gv, 3 * NF);
  zero(grbf, (long)NBPAD * NE);
  zero(gfcut, NE);
  zero(gunit, 3L * NE);

  for (int b = 1; b >= 0; --b) {
    const float* Br = msg_br + (size_t)b * F3;

    // update block backward (bf16 copies of original weights serve as Bt for g@W^T)
    k_upd_bwd1<<<NFB, 256, 0, stream>>>(gx, gv, uvb[b], vvb[b], ab[b], TA, guv, gvv);
    gemm(TA, F3, Wu2c[b], F3, TB, FD, NA, FD, F3, nullptr, 0);
    k_silu_bwd<<<NFB, 256, 0, stream>>>(TB, t2b[b], TB, NF);
    gemm(TB, FD, Wu1c[b], FD, catb, 2 * FD, NA, 2 * FD, FD, nullptr, 0);
    k_upd_bwd2<<<NFB, 256, 0, stream>>>(catb, vnb[b], vvb[b], gx, gvv);
    gemm(guv, FD, Uc[b], FD, gv, FD, 3 * NA, FD, FD, nullptr, GF_ACCUM);
    gemm(gvv, FD, Vc[b], FD, gv, FD, 3 * NA, FD, FD, nullptr, GF_ACCUM);

    // message block backward (recompute filt_raw, then edge pass)
    gemm(rbfb, NBPAD, Wrt[b], NBPAD, filtb, F3, NE, F3, NBPAD, Br, 0);
    zero(gphi, 3 * NF);
    copy(guv, gv, 3 * NF);   // snapshot of g_v' (avoid read/scatter race)
    k_edge_msg_bwd<<<EFB, 256, 0, stream>>>(phib[b], filtb, fcutb, unitb,
                                            vacc[0], eidx, gx, guv, gv, gphi,
                                            gunit, gfcut, b == 0 ? 1 : 0);
    gemm(filtb, F3, Wrp[b], F3, grbf, NBPAD, NE, NBPAD, F3, nullptr, GF_ACCUM);
    gemm(gphi, F3, W2c[b], F3, TA, FD, NA, FD, F3, nullptr, 0);
    k_silu_bwd<<<NFB, 256, 0, stream>>>(TA, t1b[b], TA, NF);
    gemm(TA, FD, W1c[b], FD, gx, FD, NA, FD, FD, nullptr, GF_ACCUM);
  }

  k_geom_bwd<<<(NE + 255) / 256, 256, 0, stream>>>(dbuf, unitb, eidx, grbf, gfcut, gunit, out + 1);
}
